// GPT_78185584656622
// MI455X (gfx1250) — matrix operand order repaired and checked by reading, not compiled, not run on hardware
//
#include <hip/hip_runtime.h>
#include <hip/hip_bf16.h>

// ---------------------------------------------------------------------------
// GPT-2 small forward on gfx1250 (CDNA5).
// All GEMMs + attention matmuls use V_WMMA_F32_16X16X32_BF16 (fp32 -> bf16
// convert on the fly, fp32 accumulate). Per-matrix weights (<=154MB) are L2
// resident (192MB), HBM sees one ~500MB pass (~21us at 23.3 TB/s).
// GEMM K-loop is register-staged double-buffered (one barrier per K tile).
// ---------------------------------------------------------------------------

typedef __bf16 v16bf __attribute__((ext_vector_type(16)));
typedef float  v8f   __attribute__((ext_vector_type(8)));

union Frag16 { v16bf v; unsigned int u[8]; };

// ---- fp32 -> bf16 (round-half-up: +0x8000, take top 16 bits) --------------
// Differs from RNE only on exact ties (~2^-16 probability), well below bf16
// quantization noise. Lowers to 2 adds + a merge (no 16-bit register moves).
__device__ inline unsigned int pack2bf(float a, float b) {
#if __has_builtin(__builtin_amdgcn_cvt_pk_bf16_f32)
  auto r = __builtin_amdgcn_cvt_pk_bf16_f32(a, b);   // v_cvt_pk_bf16_f32
  unsigned int out;
  __builtin_memcpy(&out, &r, 4);
  return out;
#else
  unsigned int ua = __float_as_uint(a) + 0x8000u;
  unsigned int ub = __float_as_uint(b) + 0x8000u;
  return (ua >> 16) | (ub & 0xFFFF0000u);
#endif
}
__device__ inline unsigned short f2bf(float f) {
  return (unsigned short)((__float_as_uint(f) + 0x8000u) >> 16);
}
__device__ inline v8f zero8() {
  v8f z;
#pragma unroll
  for (int i = 0; i < 8; ++i) z[i] = 0.0f;
  return z;
}
__device__ inline v8f wmma_bf16(v16bf a, v16bf b, v8f c) {
  // (neg_a, A, neg_b, B, c_mod, C, reuse_a, reuse_b)
  return __builtin_amdgcn_wmma_f32_16x16x32_bf16(false, a, false, b, (short)0, c,
                                                 false, false);
}
__device__ inline float gelu_tanh(float x) {
  float x3 = x * x * x;
  return 0.5f * x * (1.0f + tanhf(0.7978845608028654f * (x + 0.044715f * x3)));
}

// ---------------------------------------------------------------------------
// Generic GEMM: C[M,N] = act(A[M,K] @ B[K,N] + bias [+ residual])
// A,B fp32 in memory, staged in LDS as bf16 with double buffering.
// M multiple of 128 (M=2048 here), K multiple of 32, N arbitrary (guarded).
// Block: 256 threads (8 wave32). Tile 128x128x32. Wave tile 64x32 (4x2 WMMA).
// B staging: each thread owns (k,k+1) pairs for one column -> a single packed
// b32 LDS store into the transposed [n][k] layout (no 16-bit stores).
// ---------------------------------------------------------------------------
#define BM   128
#define BN   128
#define BKK  32
#define LDST 36   // bf16 elements per LDS row (pad; keeps 4B/8B alignment)

__global__ __launch_bounds__(256) void gemm_bf16_wmma(
    const float* __restrict__ A, const float* __restrict__ Bw,
    const float* __restrict__ bias, const float* residual,
    float* C, int M, int N, int K, int gelu)
{
  (void)M;
  __shared__ unsigned short sA[2][BM * LDST];   // [buf][row][k]
  __shared__ unsigned short sB[2][BN * LDST];   // [buf][n][k] (transposed)

  const int tid  = threadIdx.x;
  const int lane = tid & 31;
  const int wv   = tid >> 5;       // 0..7
  const int half = lane >> 4;      // 0/1
  const int lr   = lane & 15;
  const int wm   = wv >> 2;        // 0..1 -> 64 rows
  const int wn   = wv & 3;         // 0..3 -> 32 cols
  const int bm   = blockIdx.y * BM;
  const int bn   = blockIdx.x * BN;

  v8f acc[4][2];
#pragma unroll
  for (int i = 0; i < 4; ++i)
#pragma unroll
    for (int j = 0; j < 2; ++j) acc[i][j] = zero8();

  float4 ra[4];            // staged A tile (fp32)
  float  rb0[8], rb1[8];   // staged B tile: (k, k+1) pair per column

  auto load_tiles = [&](int kt) {
#pragma unroll
    for (int i = 0; i < 4; ++i) {           // A tile 128x32: 1024 float4 slots
      int s   = tid + i * 256;
      int row = s >> 3;
      int k4  = (s & 7) << 2;
      ra[i] = *(const float4*)(A + (size_t)(bm + row) * K + kt + k4);
    }
#pragma unroll
    for (int i = 0; i < 8; ++i) {           // B tile 32x128: 2048 k-pair slots
      int s  = tid + i * 256;               // 0..2047
      int n  = s & 127;
      int kp = s >> 7;                      // 0..15 -> k = 2*kp
      int gn = bn + n;
      const float* bp = Bw + (size_t)(kt + 2 * kp) * N + gn;
      bool ok = (gn < N);
      rb0[i] = ok ? bp[0] : 0.0f;           // row k
      rb1[i] = ok ? bp[N] : 0.0f;           // row k+1
    }
  };

  auto store_tiles = [&](int buf) {
#pragma unroll
    for (int i = 0; i < 4; ++i) {
      int s   = tid + i * 256;
      int row = s >> 3;
      int k4  = (s & 7) << 2;
      unsigned int* dst = (unsigned int*)&sA[buf][row * LDST + k4];
      dst[0] = pack2bf(ra[i].x, ra[i].y);
      dst[1] = pack2bf(ra[i].z, ra[i].w);
    }
#pragma unroll
    for (int i = 0; i < 8; ++i) {
      int s  = tid + i * 256;
      int n  = s & 127;
      int kp = s >> 7;
      *(unsigned int*)&sB[buf][n * LDST + 2 * kp] = pack2bf(rb0[i], rb1[i]);
    }
  };

  load_tiles(0);
  store_tiles(0);
  __syncthreads();

  int cur = 0;
  for (int kt = 0; kt < K; kt += BKK) {
    const bool has_next = (kt + BKK) < K;
    if (has_next) {
      load_tiles(kt + BKK);                 // global loads in flight over WMMA
      if (kt + 2 * BKK < K) {               // warm L2 for tile after next
        __builtin_prefetch(A + (size_t)(bm + (tid >> 1)) * K + kt + 2 * BKK + (tid & 1) * 16, 0, 3);
        __builtin_prefetch(Bw + (size_t)(kt + 2 * BKK + (tid >> 3)) * N + bn + (tid & 7) * 16, 0, 3);
      }
    }

    // ---- fragments per ISA 7.12.2 layouts, from current LDS buffer ----
    Frag16 af[4];
#pragma unroll
    for (int mt = 0; mt < 4; ++mt) {
      int row = wm * 64 + mt * 16 + lr;
      const unsigned int* rp = (const unsigned int*)&sA[cur][row * LDST];
      int kb = half << 2;   // A: half0 K0..7 & 16..23 ; half1 K8..15 & 24..31
#pragma unroll
      for (int j = 0; j < 4; ++j) {
        af[mt].u[j]     = rp[kb + j];
        af[mt].u[4 + j] = rp[8 + kb + j];
      }
    }
    Frag16 bfr[2];
#pragma unroll
    for (int nt = 0; nt < 2; ++nt) {
      int col = wn * 32 + nt * 16 + lr;
      const unsigned int* rp = (const unsigned int*)&sB[cur][col * LDST];
      int kb = half << 2;   // B: half0 K0..7 & 16..23 ; half1 K8..15 & 24..31
#pragma unroll
      for (int j = 0; j < 4; ++j) {
        bfr[nt].u[j]     = rp[kb + j];
        bfr[nt].u[4 + j] = rp[8 + kb + j];
      }
    }

#pragma unroll
    for (int mt = 0; mt < 4; ++mt)
#pragma unroll
      for (int nt = 0; nt < 2; ++nt)
        acc[mt][nt] = wmma_bf16(af[mt].v, bfr[nt].v, acc[mt][nt]);

    if (has_next) {
      store_tiles(cur ^ 1);                 // convert + stage into alt buffer
      __syncthreads();
      cur ^= 1;
    }
  }

  // ---- epilogue: bias / residual / GELU, N edge-guarded ----
#pragma unroll
  for (int mt = 0; mt < 4; ++mt) {
#pragma unroll
    for (int nt = 0; nt < 2; ++nt) {
      int col = bn + wn * 32 + nt * 16 + lr;
      if (col >= N) continue;
      float bv = bias ? bias[col] : 0.0f;
#pragma unroll
      for (int r = 0; r < 8; ++r) {
        int row  = bm + wm * 64 + mt * 16 + r + half * 8;  // C layout
        size_t o = (size_t)row * N + col;
        float v  = acc[mt][nt][r] + bv;
        if (residual) v += residual[o];
        if (gelu)     v = gelu_tanh(v);
        C[o] = v;
      }
    }
  }
}

// ---------------------------------------------------------------------------
// LayerNorm over D=768, one block (256 threads) per row.
// ---------------------------------------------------------------------------
__global__ __launch_bounds__(256) void layernorm_768(
    const float* __restrict__ x, const float* __restrict__ g,
    const float* __restrict__ b, float* __restrict__ out)
{
  const int row = blockIdx.x;
  const int tid = threadIdx.x;
  const float* xr = x + (size_t)row * 768;
  float a0 = xr[tid], a1 = xr[tid + 256], a2 = xr[tid + 512];
  float s  = a0 + a1 + a2;
  float sq = a0 * a0 + a1 * a1 + a2 * a2;
#pragma unroll
  for (int off = 16; off; off >>= 1) {
    s  += __shfl_xor(s,  off, 32);
    sq += __shfl_xor(sq, off, 32);
  }
  __shared__ float ps[8], pq[8];
  __shared__ float mv[2];
  if ((tid & 31) == 0) { ps[tid >> 5] = s; pq[tid >> 5] = sq; }
  __syncthreads();
  if (tid == 0) {
    float ts = 0.0f, tq = 0.0f;
#pragma unroll
    for (int i = 0; i < 8; ++i) { ts += ps[i]; tq += pq[i]; }
    float mean = ts * (1.0f / 768.0f);
    float var  = tq * (1.0f / 768.0f) - mean * mean;
    mv[0] = mean;
    mv[1] = rsqrtf(var + 1e-5f);
  }
  __syncthreads();
  float mean = mv[0], rstd = mv[1];
  float* orow = out + (size_t)row * 768;
  orow[tid]       = (a0 - mean) * rstd * g[tid]       + b[tid];
  orow[tid + 256] = (a1 - mean) * rstd * g[tid + 256] + b[tid + 256];
  orow[tid + 512] = (a2 - mean) * rstd * g[tid + 512] + b[tid + 512];
}

// ---------------------------------------------------------------------------
// Embedding: x[b,t,:] = wte[idx[b,t],:] + wpe[t % 1024,:]
// ---------------------------------------------------------------------------
__global__ __launch_bounds__(256) void embed_768(
    const int* __restrict__ idx, const float* __restrict__ wte,
    const float* __restrict__ wpe, float* __restrict__ x)
{
  const int t   = blockIdx.x;        // 0..B*T-1
  const int pos = t & 1023;          // T = 1024
  const int tok = idx[t];
  const float* te = wte + (size_t)tok * 768;
  const float* pe = wpe + (size_t)pos * 768;
  float* xr = x + (size_t)t * 768;
  xr[threadIdx.x]       = te[threadIdx.x]       + pe[threadIdx.x];
  xr[threadIdx.x + 256] = te[threadIdx.x + 256] + pe[threadIdx.x + 256];
  xr[threadIdx.x + 512] = te[threadIdx.x + 512] + pe[threadIdx.x + 512];
}

// ---------------------------------------------------------------------------
// Flash attention, head dim 64, causal, wave32 WMMA.
// One wave per 16 query rows per (b,h). 32 keys per iteration:
//   S (2 C-tiles) = Q(16x64) K^T  -> online softmax -> P via LDS -> O += P V.
// qkv layout: [B, T, 3*D] with q|k|v at column offsets 0|D|2D, head h at h*64.
// ---------------------------------------------------------------------------
__global__ __launch_bounds__(32) void flash_attn_64(
    const float* __restrict__ qkv, float* y)
{
  constexpr int T = 1024, D = 768, H = 12, HD = 64;
  const int lane = threadIdx.x & 31;
  const int half = lane >> 4;
  const int lr   = lane & 15;
  const int bh   = blockIdx.y;
  const int b    = bh / H, h = bh % H;
  const int q0   = blockIdx.x * 16;
  const size_t rs = 3 * D;

  __shared__ unsigned short sP[16 * LDST];

  // Q as two 16x32 A-fragments, pre-scaled by 1/sqrt(HD)
  Frag16 qf[2];
  {
    const float* qp = qkv + (size_t)(b * T + q0 + lr) * rs + h * HD;
    const float sc = 0.125f;
#pragma unroll
    for (int f = 0; f < 2; ++f) {
      int d0 = f * 32 + half * 16;
      const float2* qa = (const float2*)(qp + d0);
      const float2* qb = (const float2*)(qp + d0 + 8);
#pragma unroll
      for (int j = 0; j < 4; ++j) {
        float2 x0 = qa[j], x1 = qb[j];
        qf[f].u[j]     = pack2bf(x0.x * sc, x0.y * sc);
        qf[f].u[4 + j] = pack2bf(x1.x * sc, x1.y * sc);
      }
    }
  }

  v8f o[4];
#pragma unroll
  for (int i = 0; i < 4; ++i) o[i] = zero8();
  float m[8], l[8];
#pragma unroll
  for (int r = 0; r < 8; ++r) { m[r] = -1e30f; l[r] = 0.0f; }

  for (int kv = 0; kv < q0 + 16; kv += 32) {
    // ---- S = Q K^T : two 16x16 C-tiles over 32 keys ----
    v8f s0 = zero8(), s1 = zero8();
#pragma unroll
    for (int ct = 0; ct < 2; ++ct) {
      const float* kp = qkv + (size_t)(b * T + kv + ct * 16 + lr) * rs + D + h * HD;
#pragma unroll
      for (int f = 0; f < 2; ++f) {
        Frag16 kf;
        const float2* k2 = (const float2*)(kp + f * 32 + half * 16);
#pragma unroll
        for (int j = 0; j < 8; ++j) {
          float2 x = k2[j];
          kf.u[j] = pack2bf(x.x, x.y);
        }
        if (ct == 0) s0 = wmma_bf16(qf[f].v, kf.v, s0);
        else         s1 = wmma_bf16(qf[f].v, kf.v, s1);
      }
    }

    // ---- online softmax (rows in VGPR index, cols in lanes) ----
    const int kt0 = kv + lr, kt1 = kv + 16 + lr;
    float p0[8], p1[8];
#pragma unroll
    for (int r = 0; r < 8; ++r) {
      const int qrow = q0 + r + half * 8;
      float v0 = (kt0 <= qrow) ? s0[r] : -1e30f;
      float v1 = (kt1 <= qrow) ? s1[r] : -1e30f;
      float mx = fmaxf(v0, v1);
#pragma unroll
      for (int off = 1; off < 16; off <<= 1) mx = fmaxf(mx, __shfl_xor(mx, off, 32));
      float nm   = fmaxf(m[r], mx);
      float corr = __expf(m[r] - nm);
      m[r] = nm;
      float e0 = __expf(v0 - nm);
      float e1 = __expf(v1 - nm);
      float rsum = e0 + e1;
#pragma unroll
      for (int off = 1; off < 16; off <<= 1) rsum += __shfl_xor(rsum, off, 32);
      l[r] = l[r] * corr + rsum;
#pragma unroll
      for (int nt = 0; nt < 4; ++nt) o[nt][r] *= corr;
      p0[r] = e0; p1[r] = e1;
    }

    // ---- P: C-layout -> LDS bf16 -> A-fragment ----
#pragma unroll
    for (int r = 0; r < 8; ++r) {
      int row = r + half * 8;
      sP[row * LDST + lr]      = f2bf(p0[r]);
      sP[row * LDST + 16 + lr] = f2bf(p1[r]);
    }
    __syncthreads();
    Frag16 pf;
    {
      const unsigned int* rp = (const unsigned int*)&sP[lr * LDST];
      int kb = half << 2;
#pragma unroll
      for (int j = 0; j < 4; ++j) {
        pf.u[j]     = rp[kb + j];
        pf.u[4 + j] = rp[8 + kb + j];
      }
    }
    __syncthreads();

    // ---- O += P V ----
    const float* vb = qkv + (size_t)(b * T + kv) * rs + 2 * D + h * HD;
#pragma unroll
    for (int nt = 0; nt < 4; ++nt) {
      Frag16 vf;
      const int d  = nt * 16 + lr;
      const int k0 = half * 8;
#pragma unroll
      for (int j = 0; j < 8; ++j)
        vf.u[j] = pack2bf(vb[(size_t)(k0 + 2 * j + (j & 4) * 2) * rs + d],
                          vb[(size_t)(k0 + 2 * j + (j & 4) * 2 + 1) * rs + d]);
      o[nt] = wmma_bf16(pf.v, vf.v, o[nt]);
    }
  }

  // ---- write O / l back to y[b,t,h*64+d] ----
#pragma unroll
  for (int nt = 0; nt < 4; ++nt) {
    const int d = nt * 16 + lr;
#pragma unroll
    for (int r = 0; r < 8; ++r) {
      const int qrow = q0 + r + half * 8;
      y[(size_t)(b * T + qrow) * D + h * HD + d] = o[nt][r] / l[r];
    }
  }
}

// ---------------------------------------------------------------------------
// Host-side orchestration (graph-capture safe: launches only).
// ---------------------------------------------------------------------------
extern "C" void kernel_launch(void* const* d_in, const int* in_sizes, int n_in,
                              void* d_out, int out_size, void* d_ws, size_t ws_size,
                              hipStream_t stream) {
  (void)in_sizes; (void)n_in; (void)out_size; (void)ws_size;

  const float* wte   = (const float*)d_in[0];
  const float* wpe   = (const float*)d_in[1];
  const float* ln1g  = (const float*)d_in[2];
  const float* ln1b  = (const float*)d_in[3];
  const float* qkvw  = (const float*)d_in[4];
  const float* qkvb  = (const float*)d_in[5];
  const float* apw   = (const float*)d_in[6];
  const float* apb   = (const float*)d_in[7];
  const float* ln2g  = (const float*)d_in[8];
  const float* ln2b  = (const float*)d_in[9];
  const float* fcw   = (const float*)d_in[10];
  const float* fcb   = (const float*)d_in[11];
  const float* fpw   = (const float*)d_in[12];
  const float* fpb   = (const float*)d_in[13];
  const float* lnfg  = (const float*)d_in[14];
  const float* lnfb  = (const float*)d_in[15];
  const float* lmw   = (const float*)d_in[16];
  const int*   idx   = (const int*)d_in[17];
  float* out = (float*)d_out;

  constexpr int Bc = 2, Tc = 1024, Dc = 768, Lc = 12, Vc = 50257, Hc = 12;
  const int M = Bc * Tc;  // 2048

  // Workspace layout (~63 MB of fp32):
  float* x    = (float*)d_ws;                 // [M, D]   residual stream
  float* hbuf = x    + (size_t)M * Dc;        // [M, D]   LN output
  float* qkv  = hbuf + (size_t)M * Dc;        // [M, 3D]
  float* ybuf = qkv  + (size_t)M * 3 * Dc;    // [M, D]   attention output
  float* h2   = ybuf + (size_t)M * Dc;        // [M, 4D]  MLP hidden

  embed_768<<<M, 256, 0, stream>>>(idx, wte, wpe, x);

  for (int l = 0; l < Lc; ++l) {
    layernorm_768<<<M, 256, 0, stream>>>(x, ln1g + (size_t)l * Dc,
                                         ln1b + (size_t)l * Dc, hbuf);
    gemm_bf16_wmma<<<dim3(3 * Dc / 128, M / 128), 256, 0, stream>>>(
        hbuf, qkvw + (size_t)l * Dc * 3 * Dc, qkvb + (size_t)l * 3 * Dc,
        nullptr, qkv, M, 3 * Dc, Dc, 0);
    flash_attn_64<<<dim3(Tc / 16, Bc * Hc), 32, 0, stream>>>(qkv, ybuf);
    gemm_bf16_wmma<<<dim3(Dc / 128, M / 128), 256, 0, stream>>>(
        ybuf, apw + (size_t)l * Dc * Dc, apb + (size_t)l * Dc,
        x, x, M, Dc, Dc, 0);
    layernorm_768<<<M, 256, 0, stream>>>(x, ln2g + (size_t)l * Dc,
                                         ln2b + (size_t)l * Dc, hbuf);
    gemm_bf16_wmma<<<dim3(4 * Dc / 128, M / 128), 256, 0, stream>>>(
        hbuf, fcw + (size_t)l * Dc * 4 * Dc, fcb + (size_t)l * 4 * Dc,
        nullptr, h2, M, 4 * Dc, Dc, 1 /*gelu*/);
    gemm_bf16_wmma<<<dim3(Dc / 128, M / 128), 256, 0, stream>>>(
        h2, fpw + (size_t)l * 4 * Dc * Dc, fpb + (size_t)l * Dc,
        x, x, M, Dc, 4 * Dc, 0);
  }

  layernorm_768<<<M, 256, 0, stream>>>(x, lnfg, lnfb, hbuf);
  gemm_bf16_wmma<<<dim3((Vc + 127) / 128, M / 128), 256, 0, stream>>>(
      hbuf, lmw, nullptr, nullptr, out, M, Vc, Dc, 0);
}
